// SmartGPT_206158430286
// MI455X (gfx1250) — matrix operand order repaired and checked by reading, not compiled, not run on hardware
//
#include <hip/hip_runtime.h>
#include <hip/hip_bf16.h>
#include <math.h>

// ---------------- problem constants (match reference) ----------------
#define Vv 32000
#define Dd 768
#define Tt 1024
#define Bb 4
#define Ll 6
#define Hh 12
#define HDd 64
#define DFf 3072
#define Mrows (Bb * Tt)   // 4096

typedef __attribute__((ext_vector_type(16))) _Float16 v16h;
typedef __attribute__((ext_vector_type(8)))  _Float16 v8h;
typedef __attribute__((ext_vector_type(8)))  float    v8f;

#define WAIT_ASYNC(n) asm volatile("s_wait_asynccnt " #n ::: "memory")

__device__ __forceinline__ v8f v8f_zero() {
    v8f z;
#pragma unroll
    for (int i = 0; i < 8; ++i) z[i] = 0.0f;
    return z;
}

// A-fragment (16x32 f16, MxK): lane m (both halves row m&15) holds halves
// [cb..cb+7] and [cb+16..cb+23], cb = 0 (lanes<16) or 8 (lanes>=16).
// Caller passes p = &row[k0 + cb]; we read p[0..7] and p[16..23].
__device__ __forceinline__ v16h load_frag_A(const _Float16* __restrict__ p) {
    v8h lo = *(const v8h*)(p);
    v8h hi = *(const v8h*)(p + 16);
    v16h r;
#pragma unroll
    for (int i = 0; i < 8; ++i) { r[i] = lo[i]; r[i + 8] = hi[i]; }
    return r;
}

// B-fragment (32x16 f16, KxN): lane n = column n&15; holds K [cb..cb+7] and
// [cb+16..cb+23], cb = 0 (lanes<16) or 8 (lanes>=16), from N-major storage.
__device__ __forceinline__ v16h load_frag_B(const _Float16* __restrict__ p) {
    v8h lo = *(const v8h*)(p);
    v8h hi = *(const v8h*)(p + 16);
    v16h r;
#pragma unroll
    for (int i = 0; i < 8; ++i) { r[i] = lo[i]; r[i + 8] = hi[i]; }
    return r;
}

__device__ __forceinline__ v8f wmma16(v16h a, v16h b, v8f c) {
    return __builtin_amdgcn_wmma_f32_16x16x32_f16(false, a, false, b, (short)0, c,
                                                  false, false);
}

// ---------------- weight convert: fp32 [K,N] -> f16 [N,K] ----------------
// grid: (ceil(K/256), N); coalesced on the f16 write side.
__global__ void convert_transpose(const float* __restrict__ in,
                                  _Float16* __restrict__ out, int K, int N) {
    int n = blockIdx.y;
    int kk = blockIdx.x * 256 + threadIdx.x;
    if (kk < K)
        out[(size_t)n * K + kk] = (_Float16)in[(size_t)kk * N + n];
}

// ---------------- embedding: x = tok_emb[idx] + pos_emb ----------------
__global__ __launch_bounds__(256) void embed_kernel(const int* __restrict__ idx,
                                                    const float* __restrict__ tok,
                                                    const float* __restrict__ pos,
                                                    float* __restrict__ x) {
    int row = blockIdx.x;                 // 0..Mrows-1
    int t = row & (Tt - 1);
    const float* tr = tok + (size_t)idx[row] * Dd;
    const float* pr = pos + (size_t)t * Dd;
    float* xr = x + (size_t)row * Dd;
    for (int i = threadIdx.x; i < Dd; i += 256) xr[i] = tr[i] + pr[i];
}

// ---------------- layernorm fp32 -> f16 ----------------
__global__ __launch_bounds__(256) void layernorm_f16(const float* __restrict__ x,
                                                     const float* __restrict__ g,
                                                     const float* __restrict__ b,
                                                     _Float16* __restrict__ out) {
    int row = blockIdx.x;
    const float* xr = x + (size_t)row * Dd;
    float s = 0.f, ss = 0.f;
    for (int i = threadIdx.x; i < Dd; i += 256) {
        float v = xr[i];
        s += v; ss += v * v;
    }
#pragma unroll
    for (int off = 16; off >= 1; off >>= 1) {
        s += __shfl_xor(s, off, 32);
        ss += __shfl_xor(ss, off, 32);
    }
    __shared__ float sh0[8], sh1[8];
    int wv = threadIdx.x >> 5;
    if ((threadIdx.x & 31) == 0) { sh0[wv] = s; sh1[wv] = ss; }
    __syncthreads();
    s = 0.f; ss = 0.f;
#pragma unroll
    for (int i = 0; i < 8; ++i) { s += sh0[i]; ss += sh1[i]; }
    float mu = s * (1.0f / Dd);
    float var = ss * (1.0f / Dd) - mu * mu;
    float rstd = rsqrtf(var + 1e-5f);
    _Float16* orow = out + (size_t)row * Dd;
    for (int i = threadIdx.x; i < Dd; i += 256)
        orow[i] = (_Float16)((xr[i] - mu) * rstd * g[i] + b[i]);
}

// ---------------- LDS-staged, async double-buffered WMMA GEMM ----------------
// C[M,N] = A[M,K](f16 row-major) * Bt[N,K](f16 N-major), templated epilogue.
// MODE 0: fp32 out = acc + bias[col] (+ res); MODE 1: f16 out;
// MODE 2: f16 out, per-head transposed V ([B*H*HD, T]); MODE 3: f16 gelu(acc+bias).
enum { EP_F32_BIAS_RES = 0, EP_F16 = 1, EP_F16_VT = 2, EP_F16_GELU = 3 };

template <int MODE>
__global__ __launch_bounds__(128) void gemm_wmma(const _Float16* __restrict__ A,
                                                 const _Float16* __restrict__ Bt,
                                                 const float* __restrict__ bias,
                                                 const float* __restrict__ res,
                                                 void* __restrict__ out,
                                                 int M, int N, int K) {
    __shared__ _Float16 sA[2][64 * 32];   // 4 KB per buffer
    __shared__ _Float16 sB[2][64 * 32];

    const int tid = threadIdx.x;
    const int lane = tid & 31;
    const int wave = tid >> 5;
    const int bm = blockIdx.y * 64;
    const int bn = blockIdx.x * 64;
    const int wm = (wave >> 1) * 32;      // wave quadrant inside block tile
    const int wn = (wave & 1) * 32;
    const int lan = lane & 15;
    const int cbA = (lane < 16) ? 0 : 8;
    const int koB = (lane < 16) ? 0 : 8;

    // async staging: thread covers 16 halves (32 B) of one 64x32 tile row
    const int ra = tid >> 1;              // tile row 0..63
    const int ca = (tid & 1) * 16;        // half-chunk 0 / 16
    const _Float16* gA = A + (size_t)(bm + ra) * K + ca;
    const _Float16* gB = Bt + (size_t)(bn + ra) * K + ca;
    unsigned laddrA[2], laddrB[2];
#pragma unroll
    for (int i = 0; i < 2; ++i) {
        laddrA[i] = (unsigned)(uintptr_t)(&sA[i][ra * 32 + ca]);
        laddrB[i] = (unsigned)(uintptr_t)(&sB[i][ra * 32 + ca]);
    }

    auto issue_async = [&](int buf, int k0) {
        unsigned long long ga = (unsigned long long)(uintptr_t)(gA + k0);
        unsigned long long gb = (unsigned long long)(uintptr_t)(gB + k0);
        asm volatile(
            "global_load_async_to_lds_b128 %0, %2, off\n\t"
            "global_load_async_to_lds_b128 %0, %2, off offset:16\n\t"
            "global_load_async_to_lds_b128 %1, %3, off\n\t"
            "global_load_async_to_lds_b128 %1, %3, off offset:16"
            :
            : "v"(laddrA[buf]), "v"(laddrB[buf]), "v"(ga), "v"(gb)
            : "memory");
    };

    v8f acc[2][2];
#pragma unroll
    for (int i = 0; i < 2; ++i)
#pragma unroll
        for (int j = 0; j < 2; ++j) acc[i][j] = v8f_zero();

    const int nk = K / 32;
    issue_async(0, 0);
    for (int kb = 0; kb < nk; ++kb) {
        const int cur = kb & 1;
        if (kb + 1 < nk) {
            issue_async(cur ^ 1, (kb + 1) * 32);
            WAIT_ASYNC(0x4);              // buffer `cur` is complete (in-order)
        } else {
            WAIT_ASYNC(0x0);
        }
        __syncthreads();                  // everyone's staging is visible

        const _Float16* As = sA[cur];
        const _Float16* Bs = sB[cur];
        v16h a0 = load_frag_A(As + (wm + lan) * 32 + cbA);
        v16h a1 = load_frag_A(As + (wm + 16 + lan) * 32 + cbA);
        v16h b0 = load_frag_B(Bs + (wn + lan) * 32 + koB);
        v16h b1 = load_frag_B(Bs + (wn + 16 + lan) * 32 + koB);
        acc[0][0] = wmma16(a0, b0, acc[0][0]);
        acc[0][1] = wmma16(a0, b1, acc[0][1]);
        acc[1][0] = wmma16(a1, b0, acc[1][0]);
        acc[1][1] = wmma16(a1, b1, acc[1][1]);

        __syncthreads();                  // done reading before buffer reuse
    }

    const int hiOff = (lane < 16) ? 0 : 8;
#pragma unroll
    for (int i = 0; i < 2; ++i)
#pragma unroll
        for (int j = 0; j < 2; ++j) {
            int col = bn + wn + j * 16 + lan;
            int rb = bm + wm + i * 16 + hiOff;
            float bv = (MODE == EP_F32_BIAS_RES || MODE == EP_F16_GELU)
                           ? bias[col] : 0.0f;
#pragma unroll
            for (int r = 0; r < 8; ++r) {
                int row = rb + r;
                float v = acc[i][j][r] + bv;
                size_t oix = (size_t)row * N + col;
                if (MODE == EP_F32_BIAS_RES) {
                    if (res) v += res[oix];
                    ((float*)out)[oix] = v;
                } else if (MODE == EP_F16) {
                    ((_Float16*)out)[oix] = (_Float16)v;
                } else if (MODE == EP_F16_GELU) {
                    float gl = 0.5f * v * (1.0f + erff(v * 0.70710678118f));
                    ((_Float16*)out)[oix] = (_Float16)gl;
                } else {  // EP_F16_VT: vt[((b*H + head)*HD + d)*T + t]
                    int bz = row >> 10;       // /T
                    int t = row & (Tt - 1);
                    int head = col >> 6;      // /HD
                    int d = col & (HDd - 1);
                    ((_Float16*)out)[(((size_t)bz * Hh + head) * HDd + d) * Tt + t] =
                        (_Float16)v;
                }
            }
        }
}

// ---------------- flash attention (one wave per (b,h,16-query tile)) ---------
__global__ __launch_bounds__(128) void attention_flash(
    const _Float16* __restrict__ q, const _Float16* __restrict__ k,
    const _Float16* __restrict__ vt, _Float16* __restrict__ o) {
    __shared__ _Float16 lds[4][16 * 32];
    const int lane = threadIdx.x & 31;
    const int wave = threadIdx.x >> 5;
    const int qt = blockIdx.x * 4 + wave;   // 0..63
    const int bh = blockIdx.y;              // 0..47
    const int b = bh / Hh, h = bh % Hh;
    const int qbase = qt * 16;
    const int lan = lane & 15;
    const int cbA = (lane < 16) ? 0 : 8;
    const int koB = (lane < 16) ? 0 : 8;
    const int hiOff = (lane < 16) ? 0 : 8;

    // Q 16x64 as two A-fragments (K chunks 0..31, 32..63)
    const _Float16* qp = q + ((size_t)(b * Tt + qbase + lan)) * Dd + h * HDd + cbA;
    v16h aq0 = load_frag_A(qp);
    v16h aq1 = load_frag_A(qp + 32);

    v8f co[4];
#pragma unroll
    for (int t = 0; t < 4; ++t) co[t] = v8f_zero();
    float mrow[8], lrow[8];
#pragma unroll
    for (int r = 0; r < 8; ++r) { mrow[r] = -3.0e38f; lrow[r] = 0.0f; }

    _Float16* L = lds[wave];
    const _Float16* vbase = vt + ((size_t)bh * HDd) * Tt;

    const int nkb = (qbase + 16 + 31) / 32;  // 32-wide key blocks (causal bound)
    for (int kb = 0; kb < nkb; ++kb) {
        const int kb0 = kb * 32;
        v8f s[2];
        s[0] = v8f_zero();
        s[1] = v8f_zero();
#pragma unroll
        for (int tc = 0; tc < 2; ++tc) {
            const _Float16* kp =
                k + ((size_t)(b * Tt + kb0 + tc * 16 + lan)) * Dd + h * HDd + koB;
            v16h bk0 = load_frag_B(kp);
            v16h bk1 = load_frag_B(kp + 32);
            s[tc] = wmma16(aq0, bk0, s[tc]);
            s[tc] = wmma16(aq1, bk1, s[tc]);
        }
        // scale + causal mask
#pragma unroll
        for (int tc = 0; tc < 2; ++tc) {
            int kcol = kb0 + tc * 16 + lan;
#pragma unroll
            for (int r = 0; r < 8; ++r) {
                int row = qbase + hiOff + r;
                float v = s[tc][r] * 0.125f;  // HD^-0.5
                s[tc][r] = (kcol <= row) ? v : -3.0e38f;
            }
        }
        // online softmax (row stats live in the same lane-half as the C rows)
        float p0[8], p1[8];
#pragma unroll
        for (int r = 0; r < 8; ++r) {
            float mx = fmaxf(s[0][r], s[1][r]);
#pragma unroll
            for (int off = 8; off >= 1; off >>= 1)
                mx = fmaxf(mx, __shfl_xor(mx, off, 32));
            float mnew = fmaxf(mrow[r], mx);
            float e0 = __expf(s[0][r] - mnew);
            float e1 = __expf(s[1][r] - mnew);
            float sum = e0 + e1;
#pragma unroll
            for (int off = 8; off >= 1; off >>= 1) sum += __shfl_xor(sum, off, 32);
            float alpha = __expf(mrow[r] - mnew);
            lrow[r] = lrow[r] * alpha + sum;
            mrow[r] = mnew;
            p0[r] = e0; p1[r] = e1;
#pragma unroll
            for (int t = 0; t < 4; ++t) co[t][r] *= alpha;
        }
        // C-layout -> A-layout via per-wave LDS tile (16x32 f16)
#pragma unroll
        for (int r = 0; r < 8; ++r) {
            int row = hiOff + r;
            L[row * 32 + lan] = (_Float16)p0[r];
            L[row * 32 + 16 + lan] = (_Float16)p1[r];
        }
        asm volatile("s_wait_dscnt 0" ::: "memory");
        v16h pa = load_frag_A(L + lan * 32 + cbA);
        // O += P(16x32) * V(32x64) using Vt [B*H*HD, T]
#pragma unroll
        for (int t = 0; t < 4; ++t) {
            const _Float16* vp = vbase + ((size_t)(t * 16 + lan)) * Tt + kb0 + koB;
            v16h bv = load_frag_B(vp);
            co[t] = wmma16(pa, bv, co[t]);
        }
        asm volatile("s_wait_dscnt 0" ::: "memory");  // LDS reuse next iter
    }
    // normalize and write O (f16 row-major [B,T,D])
#pragma unroll
    for (int r = 0; r < 8; ++r) {
        float inv = 1.0f / lrow[r];
        int row = qbase + hiOff + r;
        _Float16* op = o + ((size_t)(b * Tt + row)) * Dd + h * HDd;
#pragma unroll
        for (int t = 0; t < 4; ++t)
            op[t * 16 + lan] = (_Float16)(co[t][r] * inv);
    }
}

// ---------------- host orchestration ----------------
extern "C" void kernel_launch(void* const* d_in, const int* in_sizes, int n_in,
                              void* d_out, int out_size, void* d_ws, size_t ws_size,
                              hipStream_t stream) {
    (void)in_sizes; (void)n_in; (void)out_size; (void)ws_size;
    const int*   idx   = (const int*)d_in[0];
    const float* tok   = (const float*)d_in[1];
    const float* pos   = (const float*)d_in[2];
    const float* Wq    = (const float*)d_in[3];
    const float* Wk    = (const float*)d_in[4];
    const float* Wv    = (const float*)d_in[5];
    const float* Wo    = (const float*)d_in[6];
    const float* bo    = (const float*)d_in[7];
    const float* W1    = (const float*)d_in[8];
    const float* b1    = (const float*)d_in[9];
    const float* W2    = (const float*)d_in[10];
    const float* b2    = (const float*)d_in[11];
    const float* ln1g  = (const float*)d_in[12];
    const float* ln1b  = (const float*)d_in[13];
    const float* ln2g  = (const float*)d_in[14];
    const float* ln2b  = (const float*)d_in[15];
    const float* lnfg  = (const float*)d_in[16];
    const float* lnfb  = (const float*)d_in[17];
    const float* headW = (const float*)d_in[18];
    const float* headb = (const float*)d_in[19];

    // workspace carve-up (256B aligned)
    char* p = (char*)d_ws;
    auto take = [&](size_t bytes) {
        char* r = p;
        p += (bytes + 255) & ~(size_t)255;
        return (void*)r;
    };
    const size_t DD = (size_t)Dd * Dd, DDF = (size_t)Dd * DFf;
    _Float16* wtq = (_Float16*)take(Ll * DD * 2);
    _Float16* wtk = (_Float16*)take(Ll * DD * 2);
    _Float16* wtv = (_Float16*)take(Ll * DD * 2);
    _Float16* wto = (_Float16*)take(Ll * DD * 2);
    _Float16* wt1 = (_Float16*)take(Ll * DDF * 2);
    _Float16* wt2 = (_Float16*)take(Ll * DDF * 2);
    _Float16* wth = (_Float16*)take((size_t)Dd * Vv * 2);
    float*    x   = (float*)take((size_t)Mrows * Dd * 4);
    _Float16* hb  = (_Float16*)take((size_t)Mrows * Dd * 2);
    _Float16* qb  = (_Float16*)take((size_t)Mrows * Dd * 2);
    _Float16* kb  = (_Float16*)take((size_t)Mrows * Dd * 2);
    _Float16* vtb = (_Float16*)take((size_t)Mrows * Dd * 2);
    _Float16* ob  = (_Float16*)take((size_t)Mrows * Dd * 2);
    _Float16* ff  = (_Float16*)take((size_t)Mrows * DFf * 2);

    // --- weight conversion (fp32 [K,N] -> f16 [N,K]) ---
    const dim3 cvD(Dd / 256 + (Dd % 256 != 0), Dd);      // K=768, N=768
    const dim3 cv1(Dd / 256 + (Dd % 256 != 0), DFf);     // K=768, N=3072
    const dim3 cv2(DFf / 256, Dd);                       // K=3072, N=768
    const dim3 cvH(Dd / 256 + (Dd % 256 != 0), Vv);      // K=768, N=32000
    for (int l = 0; l < Ll; ++l) {
        convert_transpose<<<cvD, 256, 0, stream>>>(Wq + l * DD, wtq + l * DD, Dd, Dd);
        convert_transpose<<<cvD, 256, 0, stream>>>(Wk + l * DD, wtk + l * DD, Dd, Dd);
        convert_transpose<<<cvD, 256, 0, stream>>>(Wv + l * DD, wtv + l * DD, Dd, Dd);
        convert_transpose<<<cvD, 256, 0, stream>>>(Wo + l * DD, wto + l * DD, Dd, Dd);
        convert_transpose<<<cv1, 256, 0, stream>>>(W1 + l * DDF, wt1 + l * DDF, Dd, DFf);
        convert_transpose<<<cv2, 256, 0, stream>>>(W2 + l * DDF, wt2 + l * DDF, DFf, Dd);
    }
    convert_transpose<<<cvH, 256, 0, stream>>>(headW, wth, Dd, Vv);

    // --- embedding ---
    embed_kernel<<<Mrows, 256, 0, stream>>>(idx, tok, pos, x);

    const dim3 blk(128);
    const dim3 gD(Dd / 64, Mrows / 64);     // N=768
    const dim3 gF(DFf / 64, Mrows / 64);    // N=3072
    const dim3 gV(Vv / 64, Mrows / 64);     // N=32000
    const dim3 gAtt(Tt / 64, Bb * Hh);      // 4 q-tiles/block x 48 (b,h)

    for (int l = 0; l < Ll; ++l) {
        layernorm_f16<<<Mrows, 256, 0, stream>>>(x, ln1g + l * Dd, ln1b + l * Dd, hb);
        gemm_wmma<EP_F16><<<gD, blk, 0, stream>>>(hb, wtq + l * DD, nullptr, nullptr, qb, Mrows, Dd, Dd);
        gemm_wmma<EP_F16><<<gD, blk, 0, stream>>>(hb, wtk + l * DD, nullptr, nullptr, kb, Mrows, Dd, Dd);
        gemm_wmma<EP_F16_VT><<<gD, blk, 0, stream>>>(hb, wtv + l * DD, nullptr, nullptr, vtb, Mrows, Dd, Dd);
        attention_flash<<<gAtt, blk, 0, stream>>>(qb, kb, vtb, ob);
        gemm_wmma<EP_F32_BIAS_RES><<<gD, blk, 0, stream>>>(ob, wto + l * DD, bo + l * Dd, x, x, Mrows, Dd, Dd);
        layernorm_f16<<<Mrows, 256, 0, stream>>>(x, ln2g + l * Dd, ln2b + l * Dd, hb);
        gemm_wmma<EP_F16_GELU><<<gF, blk, 0, stream>>>(hb, wt1 + l * DDF, b1 + l * DFf, nullptr, ff, Mrows, DFf, Dd);
        gemm_wmma<EP_F32_BIAS_RES><<<gD, blk, 0, stream>>>(ff, wt2 + l * DDF, b2 + l * Dd, x, x, Mrows, Dd, DFf);
    }
    layernorm_f16<<<Mrows, 256, 0, stream>>>(x, lnfg, lnfb, hb);
    gemm_wmma<EP_F32_BIAS_RES><<<gV, blk, 0, stream>>>(hb, wth, headb, nullptr, (float*)d_out, Mrows, Vv, Dd);
}
